// DeepseekV3FlashAttention2_50654844289108
// MI455X (gfx1250) — matrix operand order repaired and checked by reading, not compiled, not run on hardware
//
#include <hip/hip_runtime.h>
#include <hip/hip_bf16.h>

// ---------- types ----------
typedef __attribute__((ext_vector_type(16))) __bf16 v16bf;
typedef __attribute__((ext_vector_type(8)))  float  v8f;
typedef unsigned int u32x4 __attribute__((ext_vector_type(4)));
typedef int v4i __attribute__((vector_size(16)));   // matches async-LDS builtin param

union V16 {
    v16bf v;
    u32x4 u[2];
};

#define AS1 __attribute__((address_space(1)))
#define AS3 __attribute__((address_space(3)))

#if __has_builtin(__builtin_amdgcn_global_load_async_to_lds_b128)
#define HAVE_ASYNC 1
#else
#define HAVE_ASYNC 0
#endif

__device__ __forceinline__ void async_copy16(const unsigned short* g,
                                             unsigned short* l) {
#if HAVE_ASYNC
    __builtin_amdgcn_global_load_async_to_lds_b128(
        (AS1 v4i*)g, (AS3 v4i*)l, 0, 0);
#else
    *(u32x4*)l = *(const u32x4*)g;
#endif
}

__device__ __forceinline__ void wait_async0() {
#if __has_builtin(__builtin_amdgcn_s_wait_asynccnt)
    __builtin_amdgcn_s_wait_asynccnt(0);
#else
    asm volatile("s_wait_asynccnt 0x0" ::: "memory");
#endif
}

__device__ __forceinline__ unsigned short f32_to_bf16(float f) {
    unsigned int x = __float_as_uint(f);
    unsigned int r = x + 0x7fffu + ((x >> 16) & 1u);   // round-to-nearest-even
    return (unsigned short)(r >> 16);
}

// load 16 bf16 as two 16B chunks (p0, p1 must be 16B aligned)
__device__ __forceinline__ v16bf load2x16(const unsigned short* p0,
                                          const unsigned short* p1) {
    V16 t;
    t.u[0] = *(const u32x4*)p0;
    t.u[1] = *(const u32x4*)p1;
    return t.v;
}

__device__ __forceinline__ v8f wmma_bf16(v16bf a, v16bf b, v8f c) {
    // (neg_a, A, neg_b, B, c_mod, C, reuse_a, reuse_b)
    return __builtin_amdgcn_wmma_f32_16x16x32_bf16(false, a, false, b,
                                                   (short)0, c, false, false);
}

// ---------- problem constants ----------
#define S_  2048
#define D_  4096
#define H_  32
#define QR_ 1536
#define KVR_ 512
#define ROPE_ 64
#define NOPE_ 128
#define VD_ 128
#define QD_ 192

// ---------- elementwise: f32 -> bf16 ----------
__global__ __launch_bounds__(256) void cvt_f32_bf16_kernel(
    const float* __restrict__ x, unsigned short* __restrict__ y, size_t n) {
    size_t i = (size_t)blockIdx.x * blockDim.x + threadIdx.x;
    if (i < n) y[i] = f32_to_bf16(x[i]);
}

// ---------- transpose + convert: W[K][N] f32 -> Wt[N][K] bf16 ----------
__global__ __launch_bounds__(256) void tcvt_kernel(
    const float* __restrict__ W, unsigned short* __restrict__ Wt,
    int K, int N) {
    size_t i = (size_t)blockIdx.x * blockDim.x + threadIdx.x;
    size_t total = (size_t)K * N;
    if (i >= total) return;
    int n = (int)(i % N);
    size_t k = i / N;
    Wt[(size_t)n * K + k] = f32_to_bf16(W[i]);
}

// ---------- RMSNorm row + bf16 cast ----------
__global__ __launch_bounds__(256) void rmsnorm_cvt_kernel(
    const float* __restrict__ x, const float* __restrict__ w,
    unsigned short* __restrict__ out, int R, int strideIn) {
    int s = blockIdx.x;
    const float* row = x + (size_t)s * strideIn;
    float ss = 0.f;
    for (int i = threadIdx.x; i < R; i += 256) { float v = row[i]; ss += v * v; }
    __shared__ float red[8];
    for (int off = 16; off > 0; off >>= 1) ss += __shfl_xor(ss, off, 32);
    int wv = threadIdx.x >> 5, ln = threadIdx.x & 31;
    if (ln == 0) red[wv] = ss;
    __syncthreads();
    if (wv == 0) {
        float v = (ln < 8) ? red[ln] : 0.f;
        for (int off = 4; off > 0; off >>= 1) v += __shfl_xor(v, off, 32);
        if (ln == 0) red[0] = v;
    }
    __syncthreads();
    float inv = rsqrtf(red[0] / (float)R + 1e-6f);
    unsigned short* orow = out + (size_t)s * R;
    for (int i = threadIdx.x; i < R; i += 256)
        orow[i] = f32_to_bf16(row[i] * inv * w[i]);
}

// ---------- generic bf16 GEMM: C[M][N] = A[M][K] x Bt[N][K]^T ----------
// BM=128, BN=64, BK=32; 8 waves, each computing a 32x32 output quadrant.
// Double-buffered LDS, filled with async global->LDS (CDNA5 ASYNCcnt path).
__global__ __launch_bounds__(256) void gemm_bf16_kernel(
    const unsigned short* __restrict__ A,
    const unsigned short* __restrict__ Bt,
    float* __restrict__ C, int M, int N, int K) {
    __shared__ unsigned short As[2][128 * 32];   // 2 x 8 KB
    __shared__ unsigned short Bs[2][64 * 32];    // 2 x 4 KB
    int m0 = blockIdx.y * 128;
    int n0 = blockIdx.x * 64;
    int t = threadIdx.x;
    int wv = t >> 5;
    int lane = t & 31;
    int hb = lane >> 4;          // lane half
    int l15 = lane & 15;
    int wrow = (wv & 3) * 32;    // wave's 32-row strip
    int wcol = (wv >> 2) * 32;   // wave's 32-col strip

    // staging geometry: A tile 128x32 (16 elem/thread), B tile 64x32 (8 elem/thread)
    int aIdx = t * 16;
    int aRow = aIdx >> 5, aCol = aIdx & 31;
    int bIdx = t * 8;
    int bRow = bIdx >> 5, bCol = bIdx & 31;
    const unsigned short* aG = &A[(size_t)(m0 + aRow) * K + aCol];
    const unsigned short* bG = &Bt[(size_t)(n0 + bRow) * K + bCol];

    v8f acc00 = {}, acc01 = {}, acc10 = {}, acc11 = {};

    int nk = K >> 5;
    // prologue: stage k-step 0 into buffer 0
    async_copy16(aG,     &As[0][aIdx]);
    async_copy16(aG + 8, &As[0][aIdx + 8]);
    async_copy16(bG,     &Bs[0][bIdx]);

    for (int ks = 0; ks < nk; ++ks) {
        int buf = ks & 1;
        wait_async0();         // my async writes done
        __syncthreads();       // everyone's async writes visible
        if (ks + 1 < nk) {     // prefetch next step into other buffer
            const unsigned short* aN = aG + (size_t)(ks + 1) * 32;
            const unsigned short* bN = bG + (size_t)(ks + 1) * 32;
            async_copy16(aN,     &As[buf ^ 1][aIdx]);
            async_copy16(aN + 8, &As[buf ^ 1][aIdx + 8]);
            async_copy16(bN,     &Bs[buf ^ 1][bIdx]);
        }
        const unsigned short* ar0 = &As[buf][(wrow + l15) * 32];
        const unsigned short* ar1 = &As[buf][(wrow + 16 + l15) * 32];
        v16bf a0 = load2x16(ar0 + hb * 8, ar0 + hb * 8 + 16);
        v16bf a1 = load2x16(ar1 + hb * 8, ar1 + hb * 8 + 16);
        const unsigned short* br0 = &Bs[buf][(wcol + l15) * 32];
        const unsigned short* br1 = &Bs[buf][(wcol + 16 + l15) * 32];
        v16bf b0 = load2x16(br0 + hb * 8, br0 + hb * 8 + 16);
        v16bf b1 = load2x16(br1 + hb * 8, br1 + hb * 8 + 16);
        acc00 = wmma_bf16(a0, b0, acc00);
        acc01 = wmma_bf16(a0, b1, acc01);
        acc10 = wmma_bf16(a1, b0, acc10);
        acc11 = wmma_bf16(a1, b1, acc11);
    }
    int rowB = m0 + wrow + hb * 8;
    int colB = n0 + wcol + l15;
#pragma unroll
    for (int r = 0; r < 8; ++r) {
        C[(size_t)(rowB + r) * N + colB]            = acc00[r];
        C[(size_t)(rowB + r) * N + colB + 16]       = acc01[r];
        C[(size_t)(rowB + 16 + r) * N + colB]       = acc10[r];
        C[(size_t)(rowB + 16 + r) * N + colB + 16]  = acc11[r];
    }
}

// ---------- RoPE helper ----------
__device__ __forceinline__ float rope_val(const float* pe, int j, int s) {
    int fi = j & 31;
    float invf = __powf(10000.0f, -(float)fi / 32.0f);
    float ang = (float)s * invf;
    float c = __cosf(ang), sn = __sinf(ang);
    if (j < 32) return pe[2 * j] * c - pe[2 * j + 1] * sn;
    int jj = j - 32;
    return pe[2 * jj + 1] * c + pe[2 * jj] * sn;
}

// ---------- q post: q_f32[S][H*QD] -> qf bf16 [H][S][QD] with RoPE ----------
__global__ __launch_bounds__(256) void qpost_kernel(
    const float* __restrict__ q, unsigned short* __restrict__ qf) {
    size_t i = (size_t)blockIdx.x * blockDim.x + threadIdx.x;
    size_t total = (size_t)H_ * S_ * QD_;
    if (i >= total) return;
    int d = (int)(i % QD_);
    size_t hs = i / QD_;
    int s = (int)(hs % S_);
    int h = (int)(hs / S_);
    const float* qrow = q + (size_t)s * (H_ * QD_) + h * QD_;
    float val = (d < NOPE_) ? qrow[d] : rope_val(qrow + NOPE_, d - NOPE_, s);
    qf[i] = f32_to_bf16(val);
}

// ---------- kv post: kv_f32[S][H*256], ckv_f32[S][576] -> kf[H][S][192], vT[H][128][S] ----------
__global__ __launch_bounds__(256) void kvpost_kernel(
    const float* __restrict__ kv, const float* __restrict__ ckv,
    unsigned short* __restrict__ kf, unsigned short* __restrict__ vT) {
    size_t i = (size_t)blockIdx.x * blockDim.x + threadIdx.x;
    size_t total1 = (size_t)H_ * S_ * QD_;
    size_t total2 = (size_t)H_ * VD_ * S_;
    if (i < total1) {
        int d = (int)(i % QD_);
        size_t hs = i / QD_;
        int s = (int)(hs % S_);
        int h = (int)(hs / S_);
        float val;
        if (d < NOPE_) val = kv[(size_t)s * (H_ * 256) + h * 256 + d];
        else val = rope_val(ckv + (size_t)s * 576 + KVR_, d - NOPE_, s);
        kf[i] = f32_to_bf16(val);
    } else if (i < total1 + total2) {
        size_t i2 = i - total1;
        int s = (int)(i2 % S_);
        size_t hv = i2 / S_;
        int vd = (int)(hv % VD_);
        int h = (int)(hv / VD_);
        vT[i2] = f32_to_bf16(kv[(size_t)s * (H_ * 256) + h * 256 + NOPE_ + vd]);
    }
}

// ---------- flash attention (causal), bf16 WMMA, online softmax ----------
__global__ __launch_bounds__(128) void mla_attn_kernel(
    const unsigned short* __restrict__ qf,  // [H][S][192]
    const unsigned short* __restrict__ kf,  // [H][S][192]
    const unsigned short* __restrict__ vT,  // [H][128][S]
    unsigned short* __restrict__ obf) {     // [S][H*VD]
    __shared__ unsigned short pr[4][16 * 32];
    int h = blockIdx.y;
    int qb = blockIdx.x * 64;
    int wv = threadIdx.x >> 5;
    int lane = threadIdx.x & 31;
    int hb = lane >> 4;
    int l15 = lane & 15;
    int qRow0 = qb + wv * 16;

    const unsigned short* qh = qf + (size_t)h * S_ * QD_;
    const unsigned short* kh = kf + (size_t)h * S_ * QD_;
    const unsigned short* vh = vT + (size_t)h * VD_ * S_;

    // resident Q A-fragments (QD=192 -> 6 chunks of K=32)
    v16bf qa[6];
    {
        const unsigned short* qr = qh + (size_t)(qRow0 + l15) * QD_;
        int kb = hb * 8;
#pragma unroll
        for (int f = 0; f < 6; ++f)
            qa[f] = load2x16(qr + f * 32 + kb, qr + f * 32 + kb + 16);
    }

    v8f zero = {};
    v8f o[8];
#pragma unroll
    for (int t = 0; t < 8; ++t) o[t] = zero;
    float m[8], lsum[8];
#pragma unroll
    for (int r = 0; r < 8; ++r) { m[r] = -3.0e38f; lsum[r] = 0.f; }

    const float scale = 0.07216878364870322f;  // 192^-0.5
    unsigned short* myp = pr[wv];
    int kEnd = qb + 64;

    for (int kp = 0; kp < kEnd; kp += 32) {
        // ---- scores: two 16x16 tiles over 32 keys ----
        v8f s0 = zero, s1 = zero;
#pragma unroll
        for (int f = 0; f < 6; ++f) {
            const unsigned short* k0r = kh + (size_t)(kp + l15) * QD_ + f * 32 + hb * 8;
            v16bf b0 = load2x16(k0r, k0r + 16);
            s0 = wmma_bf16(qa[f], b0, s0);
            const unsigned short* k1r = kh + (size_t)(kp + 16 + l15) * QD_ + f * 32 + hb * 8;
            v16bf b1 = load2x16(k1r, k1r + 16);
            s1 = wmma_bf16(qa[f], b1, s1);
        }
        // ---- scale, causal mask, row max ----
        float mn[8];
#pragma unroll
        for (int r = 0; r < 8; ++r) {
            int row = qRow0 + r + hb * 8;
            float a0 = ((kp + l15) <= row) ? s0[r] * scale : -3.0e38f;
            float a1 = ((kp + 16 + l15) <= row) ? s1[r] * scale : -3.0e38f;
            s0[r] = a0; s1[r] = a1;
            float mx = fmaxf(a0, a1);
            mx = fmaxf(mx, __shfl_xor(mx, 1, 32));
            mx = fmaxf(mx, __shfl_xor(mx, 2, 32));
            mx = fmaxf(mx, __shfl_xor(mx, 4, 32));
            mx = fmaxf(mx, __shfl_xor(mx, 8, 32));
            mn[r] = fmaxf(m[r], mx);
        }
        // ---- exp, row sums, rescale O ----
#pragma unroll
        for (int r = 0; r < 8; ++r) {
            float esc = __expf(m[r] - mn[r]);
            m[r] = mn[r];
            float p0 = __expf(s0[r] - mn[r]);
            float p1 = __expf(s1[r] - mn[r]);
            float rs = p0 + p1;
            rs += __shfl_xor(rs, 1, 32);
            rs += __shfl_xor(rs, 2, 32);
            rs += __shfl_xor(rs, 4, 32);
            rs += __shfl_xor(rs, 8, 32);
            lsum[r] = lsum[r] * esc + rs;
            s0[r] = p0; s1[r] = p1;
#pragma unroll
            for (int t = 0; t < 8; ++t) o[t][r] *= esc;
        }
        // ---- probs C-layout -> LDS (16x32 row-major bf16) ----
#pragma unroll
        for (int r = 0; r < 8; ++r) {
            int row = r + hb * 8;
            myp[row * 32 + l15]      = f32_to_bf16(s0[r]);
            myp[row * 32 + 16 + l15] = f32_to_bf16(s1[r]);
        }
        // ---- reload probs as A-fragment, PV WMMAs ----
        {
            const unsigned short* prr = &myp[l15 * 32];
            int kb = hb * 8;
            v16bf pa = load2x16(prr + kb, prr + kb + 16);
#pragma unroll
            for (int t = 0; t < 8; ++t) {
                const unsigned short* vr = vh + (size_t)(t * 16 + l15) * S_ + kp + hb * 8;
                v16bf bv = load2x16(vr, vr + 16);
                o[t] = wmma_bf16(pa, bv, o[t]);
            }
        }
    }
    // ---- epilogue: normalize, store bf16 ----
#pragma unroll
    for (int r = 0; r < 8; ++r) {
        float inv = 1.f / lsum[r];
        int row = qRow0 + r + hb * 8;
#pragma unroll
        for (int t = 0; t < 8; ++t)
            obf[(size_t)row * (H_ * VD_) + h * VD_ + t * 16 + l15] =
                f32_to_bf16(o[t][r] * inv);
    }
}

// ---------- host orchestration ----------
extern "C" void kernel_launch(void* const* d_in, const int* in_sizes, int n_in,
                              void* d_out, int out_size, void* d_ws, size_t ws_size,
                              hipStream_t stream) {
    (void)in_sizes; (void)n_in; (void)out_size; (void)ws_size;
    const float* hidden  = (const float*)d_in[0];
    const float* Wq_a    = (const float*)d_in[1];
    const float* q_ln_w  = (const float*)d_in[2];
    const float* Wq_b    = (const float*)d_in[3];
    const float* Wkv_a   = (const float*)d_in[4];
    const float* kv_ln_w = (const float*)d_in[5];
    const float* Wkv_b   = (const float*)d_in[6];
    const float* Wo      = (const float*)d_in[7];
    float* out = (float*)d_out;

    char* ws = (char*)d_ws;
    // workspace layout (bytes), lifetime-overlapped
    const size_t OFF_W   = 0;                         // shared weight slot (33.5 MB)
    const size_t OFF_HB  = OFF_W  + (size_t)33554432; // hidden bf16, later obf (16.8 MB)
    const size_t OFF_QA  = OFF_HB + (size_t)16777216; // qa f32, later ckv f32 (12.6 MB)
    const size_t OFF_QAN = OFF_QA + (size_t)12582912; // qan bf16, later c bf16 (6.3 MB)
    const size_t OFF_BIG = OFF_QAN+ (size_t)6291456;  // q f32, later kv f32 (67.1 MB)
    const size_t OFF_QF  = OFF_BIG+ (size_t)67108864; // qf (25.2 MB)
    const size_t OFF_KF  = OFF_QF + (size_t)25165824; // kf (25.2 MB)
    const size_t OFF_VT  = OFF_KF + (size_t)25165824; // vT (16.8 MB)

    unsigned short* Wt   = (unsigned short*)(ws + OFF_W);
    unsigned short* Hb   = (unsigned short*)(ws + OFF_HB);
    float*          qa32 = (float*)(ws + OFF_QA);
    unsigned short* qan  = (unsigned short*)(ws + OFF_QAN);
    float*          big  = (float*)(ws + OFF_BIG);
    unsigned short* qfb  = (unsigned short*)(ws + OFF_QF);
    unsigned short* kfb  = (unsigned short*)(ws + OFF_KF);
    unsigned short* vtb  = (unsigned short*)(ws + OFF_VT);
    float*          ckv  = qa32;                       // reuse after qa dead
    unsigned short* cbf  = qan;                        // reuse after qan dead
    unsigned short* obf  = Hb;                         // reuse after Hb dead

    auto blocks = [](size_t n) { return (unsigned)((n + 255) / 256); };
    auto gemm = [&](const unsigned short* A, const unsigned short* Bt, float* C,
                    int M, int N, int K) {
        dim3 g((unsigned)(N / 64), (unsigned)(M / 128));
        gemm_bf16_kernel<<<g, 256, 0, stream>>>(A, Bt, C, M, N, K);
    };
    auto tcvt = [&](const float* W, int K, int N) {
        tcvt_kernel<<<blocks((size_t)K * N), 256, 0, stream>>>(W, Wt, K, N);
    };

    // 1) hidden -> bf16
    cvt_f32_bf16_kernel<<<blocks((size_t)S_ * D_), 256, 0, stream>>>(
        hidden, Hb, (size_t)S_ * D_);
    // 2) q_a = hidden @ Wq_a
    tcvt(Wq_a, D_, QR_);
    gemm(Hb, Wt, qa32, S_, QR_, D_);
    // 3) rmsnorm(q_a)
    rmsnorm_cvt_kernel<<<S_, 256, 0, stream>>>(qa32, q_ln_w, qan, QR_, QR_);
    // 4) q = q_a_norm @ Wq_b
    tcvt(Wq_b, QR_, H_ * QD_);
    gemm(qan, Wt, big, S_, H_ * QD_, QR_);
    // 5) RoPE + layout -> qf [H][S][QD]
    qpost_kernel<<<blocks((size_t)H_ * S_ * QD_), 256, 0, stream>>>(big, qfb);
    // 6) ckv = hidden @ Wkv_a
    tcvt(Wkv_a, D_, KVR_ + ROPE_);
    gemm(Hb, Wt, ckv, S_, KVR_ + ROPE_, D_);
    // 7) rmsnorm(compressed)
    rmsnorm_cvt_kernel<<<S_, 256, 0, stream>>>(ckv, kv_ln_w, cbf, KVR_, KVR_ + ROPE_);
    // 8) kv = c_norm @ Wkv_b
    tcvt(Wkv_b, KVR_, H_ * (NOPE_ + VD_));
    gemm(cbf, Wt, big, S_, H_ * (NOPE_ + VD_), KVR_);
    // 9) build kf [H][S][QD] (with RoPE'd k_pe) and vT [H][VD][S]
    kvpost_kernel<<<blocks((size_t)H_ * S_ * QD_ + (size_t)H_ * VD_ * S_),
                    256, 0, stream>>>(big, ckv, kfb, vtb);
    // 10) causal flash attention -> obf [S][H*VD]
    {
        dim3 g((unsigned)(S_ / 64), (unsigned)H_);
        mla_attn_kernel<<<g, 128, 0, stream>>>(qfb, kfb, vtb, obf);
    }
    // 11) out = o @ Wo
    tcvt(Wo, H_ * VD_, D_);
    gemm(obf, Wt, out, S_, D_, H_ * VD_);
}
